// MultiHeadSelfAttention_19258633355489
// MI455X (gfx1250) — compile-verified, matrix operand order checked by reading, not run on hardware
//
#include <hip/hip_runtime.h>

typedef __attribute__((ext_vector_type(16))) _Float16 v16h;
typedef __attribute__((ext_vector_type(8)))  _Float16 v8h;
typedef __attribute__((ext_vector_type(8)))  float    v8f;

#define DIM   768
#define HEADS 12
#define HDIM  64
#define SEQ   4096
#define BATCH 2

// D = A*B + C, 16x16x32 f16 -> f32
__device__ __forceinline__ v8f wmma32(v16h a, v16h b, v8f c) {
    return __builtin_amdgcn_wmma_f32_16x16x32_f16(false, a, false, b, (short)0, c, false, false);
}

// Load one A/B fragment (16 halves) for v_wmma_f32_16x16x32_f16 from a
// row-major row pointer. Lane layout: g = lane>>4; halves 0..7 = K g*8..g*8+7,
// halves 8..15 = K 16+g*8 .. 23+g*8.  Two 16-byte loads.
__device__ __forceinline__ v16h frag_ld(const _Float16* rowp, int g) {
    v8h lo = *(const v8h*)(rowp + g * 8);
    v8h hi = *(const v8h*)(rowp + 16 + g * 8);
    v16h f;
#pragma unroll
    for (int i = 0; i < 8; ++i) { f[i] = lo[i]; f[i + 8] = hi[i]; }
    return f;
}

// ---------------------------------------------------------------------------
// Kernel 1: qkv = x @ w_qkv + b_qkv, scattered to q/k/v f16 [B][H][N][64]
// Workgroup: 256 thr (8 waves), tile M=64 x N=128, waves 4(m) x 2(n),
// each wave: 16m x 64n = 4 C tiles.
// ---------------------------------------------------------------------------
__global__ __launch_bounds__(256) void qkv_kernel(
    const float* __restrict__ x, const float* __restrict__ w,
    const float* __restrict__ bias,
    _Float16* __restrict__ qh, _Float16* __restrict__ kh, _Float16* __restrict__ vh)
{
    __shared__ _Float16 Ash[64 * 32];    // [m][k] f16
    __shared__ _Float16 Bsh[128 * 32];   // [n][k] f16 (w transposed)

    const int nNb  = (3 * DIM) / 128;    // 18
    const int Nblk = blockIdx.x % nNb;
    const int Mblk = blockIdx.x / nNb;
    const int t    = threadIdx.x;
    const int lane = t & 31, wv = t >> 5;
    const int mw = wv & 3, nw = wv >> 1 >> 1;   // nw = wv>>2
    const int g = lane >> 4, ln = lane & 15;

    v8f acc[4];
#pragma unroll
    for (int j = 0; j < 4; ++j) acc[j] = (v8f){0,0,0,0,0,0,0,0};

    for (int k0 = 0; k0 < DIM; k0 += 32) {
        __syncthreads();
        {   // stage A: 64x32 f32 -> f16, 8 elems/thread
            int row = t >> 2, c0 = (t & 3) * 8;
            const float* src = x + (size_t)(Mblk * 64 + row) * DIM + k0 + c0;
            v8h d;
#pragma unroll
            for (int i = 0; i < 8; ++i) d[i] = (_Float16)src[i];
            *(v8h*)(Ash + row * 32 + c0) = d;
        }
        {   // stage B transposed: Bsh[n][kk] = w[k0+kk][Nblk*128+n], 16 elems/thread
            int n = t >> 1, kk0 = (t & 1) * 16;
            int col = Nblk * 128 + n;
            v8h t0, t1;
#pragma unroll
            for (int i = 0; i < 8; ++i) t0[i] = (_Float16)w[(size_t)(k0 + kk0 + i) * (3 * DIM) + col];
#pragma unroll
            for (int i = 0; i < 8; ++i) t1[i] = (_Float16)w[(size_t)(k0 + kk0 + 8 + i) * (3 * DIM) + col];
            *(v8h*)(Bsh + n * 32 + kk0)     = t0;
            *(v8h*)(Bsh + n * 32 + kk0 + 8) = t1;
        }
        __syncthreads();

        v16h a = frag_ld(Ash + (mw * 16 + ln) * 32, g);
#pragma unroll
        for (int j = 0; j < 4; ++j) {
            v16h b = frag_ld(Bsh + (nw * 64 + j * 16 + ln) * 32, g);
            acc[j] = wmma32(a, b, acc[j]);
        }
    }

    // epilogue: bias + scatter into head-major f16 q/k/v
#pragma unroll
    for (int j = 0; j < 4; ++j) {
        int cj   = Nblk * 128 + nw * 64 + j * 16 + ln;
        float bv = bias[cj];
        int tsel = cj / DIM, rem = cj % DIM;
        int h = rem / HDIM, d = rem % HDIM;
        _Float16* dst = (tsel == 0) ? qh : (tsel == 1) ? kh : vh;
#pragma unroll
        for (int r = 0; r < 8; ++r) {
            int m  = Mblk * 64 + mw * 16 + r + 8 * g;
            int b_ = m >> 12, n_ = m & 4095;
            dst[((size_t)(b_ * HEADS + h) * SEQ + n_) * HDIM + d] = (_Float16)(acc[j][r] + bv);
        }
    }
}

// ---------------------------------------------------------------------------
// Kernel 2: flash attention. 1 block = (b,h, 64 q rows), 4 waves x 16 q rows.
// Key blocks of 32; online softmax; O accumulated in f32 fragments.
// ---------------------------------------------------------------------------
__global__ __launch_bounds__(128) void attn_kernel(
    const _Float16* __restrict__ qh, const _Float16* __restrict__ kh,
    const _Float16* __restrict__ vh, _Float16* __restrict__ ah)
{
    __shared__ _Float16 Vt[64 * 32];        // V^T : [d][key] for current block
    __shared__ _Float16 Ptile[4 * 16 * 32]; // per-wave P round-trip tile

    const int bh   = blockIdx.x >> 6;       // 0..23
    const int qblk = blockIdx.x & 63;
    const int b = bh / HEADS, h = bh % HEADS;
    const int t = threadIdx.x, lane = t & 31, wv = t >> 5;
    const int g = lane >> 4, ln = lane & 15;
    const float scale = 0.125f;             // 1/sqrt(64)

    const _Float16* Kbase = kh + (size_t)bh * SEQ * HDIM;
    const _Float16* Vbase = vh + (size_t)bh * SEQ * HDIM;
    _Float16* Pl = Ptile + wv * 16 * 32;

    // Q fragments for this wave's 16 rows (d split into two K=32 chunks)
    const _Float16* Qp = qh + ((size_t)bh * SEQ + qblk * 64 + wv * 16 + ln) * HDIM;
    v16h q0 = frag_ld(Qp, g);
    v16h q1 = frag_ld(Qp + 32, g);

    v8f O[4];
#pragma unroll
    for (int c = 0; c < 4; ++c) O[c] = (v8f){0,0,0,0,0,0,0,0};
    float mrow[8], lrow[8];
#pragma unroll
    for (int r = 0; r < 8; ++r) { mrow[r] = -1e30f; lrow[r] = 0.0f; }

    for (int kb = 0; kb < SEQ; kb += 32) {
        __syncthreads();
        {   // cooperative stage of V block transposed: Vt[d][key]
            int row = t >> 2, c0 = (t & 3) * 16;
            const _Float16* src = Vbase + (size_t)(kb + row) * HDIM + c0;
            v8h a0 = *(const v8h*)src;
            v8h a1 = *(const v8h*)(src + 8);
#pragma unroll
            for (int i = 0; i < 8; ++i) {
                Vt[(c0 + i) * 32 + row]     = a0[i];
                Vt[(c0 + 8 + i) * 32 + row] = a1[i];
            }
        }
        __syncthreads();

        // S = Q @ K^T  (two 16-key subtiles, K rows read directly from global)
        v8f S0 = (v8f){0,0,0,0,0,0,0,0}, S1 = S0;
        const _Float16* Kr0 = Kbase + (size_t)(kb + ln) * HDIM;
        const _Float16* Kr1 = Kbase + (size_t)(kb + 16 + ln) * HDIM;
        S0 = wmma32(q0, frag_ld(Kr0, g), S0);
        S0 = wmma32(q1, frag_ld(Kr0 + 32, g), S0);
        S1 = wmma32(q0, frag_ld(Kr1, g), S1);
        S1 = wmma32(q1, frag_ld(Kr1 + 32, g), S1);

        // online softmax; row m = r + 8g lives in the 16 lanes of group g
#pragma unroll
        for (int r = 0; r < 8; ++r) {
            float s0 = S0[r] * scale, s1 = S1[r] * scale;
            float mx = fmaxf(s0, s1);
#pragma unroll
            for (int msk = 1; msk < 16; msk <<= 1) mx = fmaxf(mx, __shfl_xor(mx, msk, 32));
            float mnew = fmaxf(mrow[r], mx);
            float corr = __expf(mrow[r] - mnew);
            float p0 = __expf(s0 - mnew), p1 = __expf(s1 - mnew);
            float ps = p0 + p1;
#pragma unroll
            for (int msk = 1; msk < 16; msk <<= 1) ps += __shfl_xor(ps, msk, 32);
            lrow[r] = lrow[r] * corr + ps;
            mrow[r] = mnew;
#pragma unroll
            for (int c = 0; c < 4; ++c) O[c][r] *= corr;
            int m = r + 8 * g;
            Pl[m * 32 + ln]      = (_Float16)p0;
            Pl[m * 32 + 16 + ln] = (_Float16)p1;
        }

        // P back as an A fragment (same-wave LDS: DS ops stay in order)
        v16h pf = frag_ld(Pl + ln * 32, g);
#pragma unroll
        for (int c = 0; c < 4; ++c) {
            v16h vf = frag_ld(Vt + (c * 16 + ln) * 32, g);
            O[c] = wmma32(pf, vf, O[c]);
        }
    }

    // normalize and store attn-out as f16 [B][N][DIM] (col = h*64 + d)
    int qrowbase = qblk * 64 + wv * 16;
#pragma unroll
    for (int r = 0; r < 8; ++r) {
        float inv = 1.0f / lrow[r];
        int qrow = qrowbase + r + 8 * g;
        size_t rowoff = ((size_t)b * SEQ + qrow) * DIM + h * HDIM;
#pragma unroll
        for (int c = 0; c < 4; ++c)
            ah[rowoff + c * 16 + ln] = (_Float16)(O[c][r] * inv);
    }
}

// ---------------------------------------------------------------------------
// Kernel 3: out = attn @ w_proj + b_proj  (f16 A, f32 B staged to f16, f32 out)
// ---------------------------------------------------------------------------
__global__ __launch_bounds__(256) void proj_kernel(
    const _Float16* __restrict__ a, const float* __restrict__ w,
    const float* __restrict__ bias, float* __restrict__ out)
{
    __shared__ _Float16 Ash[64 * 32];
    __shared__ _Float16 Bsh[128 * 32];

    const int nNb  = DIM / 128;          // 6
    const int Nblk = blockIdx.x % nNb;
    const int Mblk = blockIdx.x / nNb;
    const int t    = threadIdx.x;
    const int lane = t & 31, wv = t >> 5;
    const int mw = wv & 3, nw = wv >> 2;
    const int g = lane >> 4, ln = lane & 15;

    v8f acc[4];
#pragma unroll
    for (int j = 0; j < 4; ++j) acc[j] = (v8f){0,0,0,0,0,0,0,0};

    for (int k0 = 0; k0 < DIM; k0 += 32) {
        __syncthreads();
        {   // stage A (already f16): straight 16B copy
            int row = t >> 2, c0 = (t & 3) * 8;
            *(v8h*)(Ash + row * 32 + c0) =
                *(const v8h*)(a + (size_t)(Mblk * 64 + row) * DIM + k0 + c0);
        }
        {   // stage B transposed
            int n = t >> 1, kk0 = (t & 1) * 16;
            int col = Nblk * 128 + n;
            v8h t0, t1;
#pragma unroll
            for (int i = 0; i < 8; ++i) t0[i] = (_Float16)w[(size_t)(k0 + kk0 + i) * DIM + col];
#pragma unroll
            for (int i = 0; i < 8; ++i) t1[i] = (_Float16)w[(size_t)(k0 + kk0 + 8 + i) * DIM + col];
            *(v8h*)(Bsh + n * 32 + kk0)     = t0;
            *(v8h*)(Bsh + n * 32 + kk0 + 8) = t1;
        }
        __syncthreads();

        v16h af = frag_ld(Ash + (mw * 16 + ln) * 32, g);
#pragma unroll
        for (int j = 0; j < 4; ++j) {
            v16h bf = frag_ld(Bsh + (nw * 64 + j * 16 + ln) * 32, g);
            acc[j] = wmma32(af, bf, acc[j]);
        }
    }

#pragma unroll
    for (int j = 0; j < 4; ++j) {
        int cj   = Nblk * 128 + nw * 64 + j * 16 + ln;
        float bv = bias[cj];
#pragma unroll
        for (int r = 0; r < 8; ++r) {
            int m = Mblk * 64 + mw * 16 + r + 8 * g;
            out[(size_t)m * DIM + cj] = acc[j][r] + bv;
        }
    }
}

// ---------------------------------------------------------------------------
extern "C" void kernel_launch(void* const* d_in, const int* in_sizes, int n_in,
                              void* d_out, int out_size, void* d_ws, size_t ws_size,
                              hipStream_t stream) {
    const float* x      = (const float*)d_in[0];
    const float* w_qkv  = (const float*)d_in[1];
    const float* b_qkv  = (const float*)d_in[2];
    const float* w_proj = (const float*)d_in[3];
    const float* b_proj = (const float*)d_in[4];
    float* out = (float*)d_out;

    // workspace: q,k,v f16 head-major + attn-out f16 row-major  (~48 MB)
    _Float16* ws = (_Float16*)d_ws;
    const size_t per = (size_t)BATCH * HEADS * SEQ * HDIM;   // 6291456 halves
    _Float16* qh = ws;
    _Float16* kh = qh + per;
    _Float16* vh = kh + per;
    _Float16* ah = vh + per;                                  // [8192][768]

    // 1) QKV projection: M blocks 8192/64=128, N blocks 2304/128=18
    qkv_kernel<<<128 * 18, 256, 0, stream>>>(x, w_qkv, b_qkv, qh, kh, vh);
    // 2) flash attention: B*H*(4096/64) = 1536 blocks of 4 waves
    attn_kernel<<<BATCH * HEADS * (SEQ / 64), 128, 0, stream>>>(qh, kh, vh, ah);
    // 3) output projection: 128 * (768/128=6) blocks
    proj_kernel<<<128 * 6, 256, 0, stream>>>(ah, w_proj, b_proj, out);
}
